// LocationSensitiveAttention_18056042512708
// MI455X (gfx1250) — matrix operand order repaired and checked by reading, not compiled, not run on hardware
//
#include <hip/hip_runtime.h>
#include <hip/hip_bf16.h>
#include <math.h>

typedef _Float16 v16h __attribute__((ext_vector_type(16)));
typedef _Float16 v8h  __attribute__((ext_vector_type(8)));
typedef float    v8f  __attribute__((ext_vector_type(8)));

#define T_LEN 2048
#define B_SZ  256
#define H_SZ  256
#define QH_SZ 512
#define NF    32
#define KW    31
#define PADW  15
#define NSPLIT 16

// workspace byte offsets
#define OFF_Q    0u          // B*H  f32 = 262144
#define OFF_LOC  262144u     // B*T*NF f16 = 33554432
#define OFF_GATE 33816576u   // B*T  f32 = 2097152
#define OFF_CUM  35913728u   // B*T  f32 = 2097152
#define OFF_PART 38010880u   // NSPLIT*B*H f32 = 4194304

// Branch-free tanh: prefer the gfx1250 V_TANH_F32 trans op if clang exposes it,
// else 1 - 2*rcp(exp(2x)+1) (exp overflow/underflow saturates correctly, no branches).
__device__ __forceinline__ float fast_tanh(float x) {
#if __has_builtin(__builtin_amdgcn_tanhf)
  return __builtin_amdgcn_tanhf(x);
#else
  const float e = __expf(2.f * x);
  return 1.f - 2.f * __builtin_amdgcn_rcpf(e + 1.f);
#endif
}

__device__ __forceinline__ float fast_sigmoid(float x) {
  return __builtin_amdgcn_rcpf(1.f + __expf(-x));
}

// ---------------- Kernel 1: q = (relu(query*Wq1^T+bq1))*Wq2^T + bq2 ----------
__global__ __launch_bounds__(256) void k_query(
    const float* __restrict__ query, const float* __restrict__ Wq1,
    const float* __restrict__ bq1, const float* __restrict__ Wq2,
    const float* __restrict__ bq2, float* __restrict__ qout) {
  __shared__ float qv[QH_SZ];
  __shared__ float hid[H_SZ];
  const int b = blockIdx.x, tid = threadIdx.x;
  qv[tid]       = query[(size_t)b * QH_SZ + tid];
  qv[tid + 256] = query[(size_t)b * QH_SZ + tid + 256];
  __syncthreads();
  float acc = bq1[tid];
  const float* wr = Wq1 + (size_t)tid * QH_SZ;
  #pragma unroll 8
  for (int i = 0; i < QH_SZ; ++i) acc = fmaf(qv[i], wr[i], acc);
  hid[tid] = fmaxf(acc, 0.f);
  __syncthreads();
  float a2 = bq2[tid];
  const float* w2 = Wq2 + (size_t)tid * H_SZ;
  #pragma unroll 8
  for (int i = 0; i < H_SZ; ++i) a2 = fmaf(hid[i], w2[i], a2);
  qout[(size_t)b * H_SZ + tid] = a2;
}

// ---------------- Kernel 2: masked-pad conv, gate sigmoid, masked cum --------
__global__ __launch_bounds__(256) void k_conv(
    const float* __restrict__ cum, const unsigned char* __restrict__ mask,
    const float* __restrict__ init, const float* __restrict__ conv_w,
    const float* __restrict__ conv_b, _Float16* __restrict__ loc,
    float* __restrict__ gate_sig, float* __restrict__ cum_m) {
  __shared__ float w[(NF + 1) * KW];
  __shared__ float bb[NF + 1];
  const int tid = threadIdx.x;
  for (int i = tid; i < (NF + 1) * KW; i += 256) w[i] = conv_w[i];
  if (tid < NF + 1) bb[tid] = conv_b[tid];
  __syncthreads();
  const int gidx = blockIdx.x * 256 + tid;
  const int b = gidx / T_LEN, t = gidx % T_LEN;
  const float ini = init[b];
  float acc[NF + 1];
  #pragma unroll
  for (int c = 0; c <= NF; ++c) acc[c] = bb[c];
  for (int k = 0; k < KW; ++k) {
    const int idx = t + k;
    float xv;
    if (idx < PADW) {
      xv = ini;
    } else {
      const int tt = idx - PADW;
      xv = (tt < T_LEN && mask[(size_t)b * T_LEN + tt]) ? cum[(size_t)b * T_LEN + tt] : 0.f;
    }
    #pragma unroll
    for (int c = 0; c <= NF; ++c) acc[c] = fmaf(w[c * KW + k], xv, acc[c]);
  }
  _Float16* lp = loc + (size_t)gidx * NF;
  #pragma unroll
  for (int c = 0; c < NF; ++c) lp[c] = (_Float16)acc[c];
  gate_sig[gidx] = fast_sigmoid(acc[NF]);
  cum_m[gidx] = mask[gidx] ? cum[gidx] : 0.f;
}

// ---------------- Kernel 3: WMMA score + tanh + dot(W_score) + softmax -------
__global__ __launch_bounds__(256) void k_score(
    const _Float16* __restrict__ loc, const float* __restrict__ W_align,
    const float* __restrict__ W_score, const float* __restrict__ qbuf,
    const unsigned char* __restrict__ mask, const float* __restrict__ cum_m,
    const float* __restrict__ gate_sig,
    float* __restrict__ out_cumulative, float* __restrict__ out_alignment) {
  __shared__ _Float16 wal[H_SZ * NF];   // 16 KB: W_align in f16, [h][f]
  __shared__ float q_l[H_SZ];
  __shared__ float wsc[H_SZ];
  __shared__ float s_l[T_LEN];          // 8 KB
  __shared__ float red[16];
  const int b = blockIdx.x, tid = threadIdx.x;
  for (int i = tid; i < H_SZ * NF; i += 256) wal[i] = (_Float16)W_align[i];
  q_l[tid] = qbuf[(size_t)b * H_SZ + tid];
  wsc[tid] = W_score[tid];
  __syncthreads();

  const int lane = tid & 31, wid = tid >> 5;
  const int sel = lane >> 4;       // half-wave select
  const int n   = lane & 15;       // row / column index within 16

  // Per-lane loop invariants: q[h] and wscore[h] for this lane's 16 columns.
  float qreg[16], wreg[16];
  #pragma unroll
  for (int ht = 0; ht < 16; ++ht) {
    qreg[ht] = q_l[ht * 16 + n];
    wreg[ht] = wsc[ht * 16 + n];
  }

  // 128 t-tiles of 16; each of 8 waves handles 16 tiles
  for (int tt = wid; tt < T_LEN / 16; tt += 8) {
    const int tbase = tt * 16;
    // A tile: row m = n (t = tbase+n); lane<16: K {0..7,16..23}; lane>=16: K {8..15,24..31}
    const _Float16* lp = loc + ((size_t)b * T_LEN + tbase + n) * NF;
    v8h a0 = *(const v8h*)(lp + sel * 8);
    v8h a1 = *(const v8h*)(lp + 16 + sel * 8);
    v16h A = __builtin_shufflevector(a0, a1, 0, 1, 2, 3, 4, 5, 6, 7,
                                             8, 9, 10, 11, 12, 13, 14, 15);
    float sacc[8];
    #pragma unroll
    for (int j = 0; j < 8; ++j) sacc[j] = 0.f;

    for (int ht = 0; ht < 16; ++ht) {
      const int h = ht * 16 + n;     // this lane's column
      // B tile: column h, lane<16: K {0..7,16..23}; lane>=16: K {8..15,24..31}
      v8h b0 = *(const v8h*)(&wal[h * NF + sel * 8]);
      v8h b1 = *(const v8h*)(&wal[h * NF + 16 + sel * 8]);
      v16h Bm = __builtin_shufflevector(b0, b1, 0, 1, 2, 3, 4, 5, 6, 7,
                                                8, 9, 10, 11, 12, 13, 14, 15);
      v8f c = {};
      c = __builtin_amdgcn_wmma_f32_16x16x32_f16(false, A, false, Bm,
                                                 (short)0, c, false, false);
      const float qh = qreg[ht], wh = wreg[ht];
      #pragma unroll
      for (int j = 0; j < 8; ++j)
        sacc[j] = fmaf(fast_tanh(c[j] + qh), wh, sacc[j]);
    }
    // reduce over N (16 lanes of each half-wave); VGPR j holds row j + 8*sel
    #pragma unroll
    for (int j = 0; j < 8; ++j) {
      float v = sacc[j];
      v += __shfl_xor(v, 1, 16);
      v += __shfl_xor(v, 2, 16);
      v += __shfl_xor(v, 4, 16);
      v += __shfl_xor(v, 8, 16);
      sacc[j] = v;
    }
    if (n == 0) {
      #pragma unroll
      for (int j = 0; j < 8; ++j) s_l[tbase + sel * 8 + j] = sacc[j];
    }
  }
  __syncthreads();

  // masked softmax over T=2048 (each thread owns 8 t's)
  float sv[8], ev[8];
  float lmax = -INFINITY;
  #pragma unroll
  for (int r = 0; r < 8; ++r) {
    const int t = r * 256 + tid;
    const float s = mask[(size_t)b * T_LEN + t] ? s_l[t] : -INFINITY;
    sv[r] = s;
    lmax = fmaxf(lmax, s);
  }
  for (int m = 16; m >= 1; m >>= 1) lmax = fmaxf(lmax, __shfl_xor(lmax, m, 32));
  if (lane == 0) red[wid] = lmax;
  __syncthreads();
  float bmax = red[0];
  #pragma unroll
  for (int i = 1; i < 8; ++i) bmax = fmaxf(bmax, red[i]);
  float lsum = 0.f;
  #pragma unroll
  for (int r = 0; r < 8; ++r) { ev[r] = __expf(sv[r] - bmax); lsum += ev[r]; }
  for (int m = 16; m >= 1; m >>= 1) lsum += __shfl_xor(lsum, m, 32);
  if (lane == 0) red[8 + wid] = lsum;   // disjoint slots, no extra sync needed
  __syncthreads();
  float bsum = 0.f;
  #pragma unroll
  for (int i = 0; i < 8; ++i) bsum += red[8 + i];
  const float inv = __builtin_amdgcn_rcpf(bsum);
  #pragma unroll
  for (int r = 0; r < 8; ++r) {
    const int t = r * 256 + tid;
    const size_t gi = (size_t)b * T_LEN + t;
    const float a = ev[r] * inv;
    out_alignment[gi] = a;
    out_cumulative[gi] = cum_m[gi] + a * gate_sig[gi];
  }
}

// ---------------- Kernel 4: context partial sums (bandwidth kernel) ----------
__global__ __launch_bounds__(256) void k_ctx_part(
    const float* __restrict__ align, const float* __restrict__ enc,
    float* __restrict__ part) {
  __shared__ float a_l[T_LEN / NSPLIT];
  const int split = blockIdx.x, b = blockIdx.y, tid = threadIdx.x;
  const int TC = T_LEN / NSPLIT;           // 128
  const int t0 = split * TC;
  if (tid < TC) a_l[tid] = align[(size_t)b * T_LEN + t0 + tid];
  __syncthreads();
  float acc = 0.f;
  #pragma unroll 4
  for (int i = 0; i < TC; ++i) {
    const int t = t0 + i;
    acc = fmaf(a_l[i], enc[((size_t)t * B_SZ + b) * H_SZ + tid], acc);
  }
  part[((size_t)split * B_SZ + b) * H_SZ + tid] = acc;
}

// ---------------- Kernel 5: reduce partials -> context -----------------------
__global__ __launch_bounds__(256) void k_ctx_reduce(
    const float* __restrict__ part, float* __restrict__ ctx) {
  const int b = blockIdx.x, tid = threadIdx.x;
  float acc = 0.f;
  #pragma unroll
  for (int s = 0; s < NSPLIT; ++s)
    acc += part[((size_t)s * B_SZ + b) * H_SZ + tid];
  ctx[(size_t)b * H_SZ + tid] = acc;
}

extern "C" void kernel_launch(void* const* d_in, const int* in_sizes, int n_in,
                              void* d_out, int out_size, void* d_ws, size_t ws_size,
                              hipStream_t stream) {
  (void)in_sizes; (void)n_in; (void)out_size; (void)ws_size;
  const float*         enc     = (const float*)d_in[0];
  const unsigned char* mask    = (const unsigned char*)d_in[1];  // jnp bool -> 1 byte
  const float*         query   = (const float*)d_in[2];
  const float*         cum     = (const float*)d_in[3];
  const float*         init    = (const float*)d_in[4];
  const float*         conv_w  = (const float*)d_in[5];
  const float*         conv_b  = (const float*)d_in[6];
  const float*         Wq1     = (const float*)d_in[7];
  const float*         bq1     = (const float*)d_in[8];
  const float*         Wq2     = (const float*)d_in[9];
  const float*         bq2     = (const float*)d_in[10];
  const float*         W_align = (const float*)d_in[11];
  const float*         W_score = (const float*)d_in[12];

  float* out       = (float*)d_out;
  float* out_ctx   = out;                                  // (B,H)
  float* out_cum   = out + B_SZ * H_SZ;                    // (B,T)
  float* out_align = out + B_SZ * H_SZ + B_SZ * T_LEN;     // (B,T)

  char* ws = (char*)d_ws;
  float*    qbuf  = (float*)(ws + OFF_Q);
  _Float16* loc   = (_Float16*)(ws + OFF_LOC);
  float*    gate  = (float*)(ws + OFF_GATE);
  float*    cum_m = (float*)(ws + OFF_CUM);
  float*    part  = (float*)(ws + OFF_PART);

  k_query<<<B_SZ, 256, 0, stream>>>(query, Wq1, bq1, Wq2, bq2, qbuf);
  k_conv<<<(B_SZ * T_LEN) / 256, 256, 0, stream>>>(cum, mask, init, conv_w, conv_b,
                                                   loc, gate, cum_m);
  k_score<<<B_SZ, 256, 0, stream>>>(loc, W_align, W_score, qbuf, mask, cum_m, gate,
                                    out_cum, out_align);
  k_ctx_part<<<dim3(NSPLIT, B_SZ), 256, 0, stream>>>(out_align, enc, part);
  k_ctx_reduce<<<B_SZ, 256, 0, stream>>>(part, out_ctx);
}
